// RAFTStereo_3350074491489
// MI455X (gfx1250) — matrix operand order repaired and checked by reading, not compiled, not run on hardware
//
#include <hip/hip_runtime.h>
#include <math.h>

typedef __attribute__((ext_vector_type(16))) _Float16 v16h;
typedef __attribute__((ext_vector_type(8)))  _Float16 v8h;
typedef __attribute__((ext_vector_type(8)))  float    v8f;

#define N_ 2
#define C_ 256
#define H_ 96
#define W_ 240
#define HID_ 128
#define ITERS_ 12
#define CORRCH_ 36
#define HW_ ((size_t)H_*(size_t)W_)

static __device__ __forceinline__ v8f wmma_f16(v16h a, v16h b, v8f c) {
  return __builtin_amdgcn_wmma_f32_16x16x32_f16(false, a, false, b, (short)0, c, false, false);
}

// ---------------------------------------------------------------------------
// Weight pre-pack into WMMA A-fragment order:
// packed[((t*chunks+ck)*KHW+q)*512 + lane*16 + e]
// lane<16: e 0..7 -> K 0..7, e 8..15 -> K 16..23 ; lane>=16: K 8..15 / 24..31
// ---------------------------------------------------------------------------
__global__ void pack_w_k(const float* __restrict__ w, _Float16* __restrict__ pk,
                         int Cout, int Cin, int KHW, int total) {
  int i = blockIdx.x * blockDim.x + threadIdx.x;
  if (i >= total) return;
  int e = i & 15;
  int l = (i >> 4) & 31;
  int rest = i >> 9;
  int q = rest % KHW;
  int rest2 = rest / KHW;
  int chunks = (Cin + 31) >> 5;
  int ck = rest2 % chunks;
  int t  = rest2 / chunks;
  int M = t * 16 + (l & 15);
  int kk = (l < 16) ? ((e < 8) ? e : e + 8) : ((e < 8) ? e + 8 : e + 16);
  int k = ck * 32 + kk;
  float v = 0.f;
  if (M < Cout && k < Cin) v = w[((size_t)M * Cin + k) * KHW + q];
  pk[i] = (_Float16)v;
}

// ---------------------------------------------------------------------------
// Generic implicit-GEMM conv, NCHW fp32 in/out, f16 WMMA inside.
// Block: 256 thr = 8 waves.
//   PIXT=128: tile 32 Cout x 128 px; each wave: 1 pixel subtile, 2 Cout tiles
//             (B fragment reused by both WMMAs).
//   PIXT=64 : tile 32 Cout x 64 px; each wave: 1 pixel subtile, 1 Cout tile.
// inA/inB: channel-split input (CinA multiple of 32 when inB used).
// ACT: 0 none, 1 relu, 2 sigmoid, 3 tanh.  out = scale * act(conv + bias + addt)
// ---------------------------------------------------------------------------
template<int KH, int KW, int ACT, int PIXT>
__global__ __launch_bounds__(256) void conv_wmma_k(
    const float* __restrict__ inA, int CinA,
    const float* __restrict__ inB, int Cin,
    const _Float16* __restrict__ wpk,
    const float* __restrict__ bias,
    const float* __restrict__ addt,
    float* __restrict__ out,
    int Cout, int outCO, int outC, float scale)
{
  constexpr int PAD  = KH / 2;
  constexpr int XW   = PIXT + 2 * PAD;
  constexpr int WPP  = PIXT / 16;      // waves along pixel axis (8 or 4)
  constexpr int ACCS = WPP / 4;        // cout tiles per wave (2 or 1)
  constexpr int SLAB = KH * XW * 32;
  __shared__ alignas(32) _Float16 slab[SLAB];

  const int cbCount = (Cout + 31) >> 5;
  const int tiles   = (Cout + 15) >> 4;
  const int n  = blockIdx.z / cbCount;
  const int cb = blockIdx.z % cbCount;
  const int h  = blockIdx.y;
  const int wT = blockIdx.x * PIXT;
  const int tid = threadIdx.x, lane = tid & 31, wave = tid >> 5;
  const int pixSub   = wave % WPP;
  const int coutHalf = wave / WPP;     // 0 when ACCS==2
  const int tile0 = cb * 2 + ((ACCS == 2) ? 0 : coutHalf);
  const int chunks = (Cin + 31) >> 5;
  const int CinB = Cin - CinA;

  v8f acc[ACCS];
  #pragma unroll
  for (int a = 0; a < ACCS; a++) acc[a] = (v8f){0.f,0.f,0.f,0.f,0.f,0.f,0.f,0.f};

  for (int ck = 0; ck < chunks; ck++) {
    __syncthreads();
    // stage 32-ch slab as f16; 4 channels packed per ds_store_b64.
    // Predicates hoisted to quad granularity: the common path is 4 clean
    // global_load_b32 + cvt under a single branch.
    for (int i = tid; i < SLAB / 4; i += 256) {
      int c4 = (i & 7) * 4;
      int rx = i >> 3;
      int x  = rx % XW;
      int r  = rx / XW;
      int gw = wT + x - PAD;
      int gh = h  + r - PAD;
      bool inb = ((unsigned)gw < (unsigned)W_) && ((unsigned)gh < (unsigned)H_);
      int gc0 = ck * 32 + c4;
      union { _Float16 hh[4]; unsigned long long u; } p;
      p.u = 0ull;
      if (inb && gc0 < Cin) {
        // channel quad lies entirely in inA or entirely in inB (CinA % 32 == 0)
        const float* src;
        size_t base;
        if (gc0 < CinA) { src = inA; base = ((size_t)n * CinA + gc0) * HW_; }
        else            { src = inB; base = ((size_t)n * CinB + (gc0 - CinA)) * HW_; }
        base += (size_t)gh * W_ + gw;
        if (gc0 + 4 <= Cin) {
          #pragma unroll
          for (int j = 0; j < 4; j++) p.hh[j] = (_Float16)src[base + (size_t)j * HW_];
        } else {
          #pragma unroll
          for (int j = 0; j < 4; j++)
            if (gc0 + j < Cin) p.hh[j] = (_Float16)src[base + (size_t)j * HW_];
        }
      }
      *(unsigned long long*)&slab[(size_t)(r * XW + x) * 32 + c4] = p.u;
    }
    __syncthreads();

    const _Float16* wb0 =
        wpk + ((((size_t)tile0 * chunks + ck) * (KH * KW)) * 32 + lane) * 16;
    const _Float16* wb1 = wb0 + (size_t)chunks * (KH * KW) * 512;
    if (ck + 1 < chunks) __builtin_prefetch(wb0 + (size_t)(KH * KW) * 512, 0, 1);
    const int cb16  = (lane < 16) ? 0 : 8;
    const int xbase = pixSub * 16 + (lane & 15);
    const bool t0ok = tile0 < tiles;
    const bool t1ok = (ACCS == 2) && (tile0 + 1 < tiles);
    #pragma unroll
    for (int q = 0; q < KH * KW; q++) {
      const int ky = q / KW, kx = q % KW;
      union { v16h v; v8h h2[2]; } bu;
      bu.h2[0] = *(const v8h*)&slab[(size_t)(ky * XW + xbase + kx) * 32 + cb16];
      bu.h2[1] = *(const v8h*)&slab[(size_t)(ky * XW + xbase + kx) * 32 + cb16 + 16];
      v16h b = bu.v;
      if (t0ok) {
        v16h a0 = *(const v16h*)(wb0 + (size_t)q * 512);
        acc[0] = wmma_f16(a0, b, acc[0]);
      }
      if (ACCS == 2 && t1ok) {
        v16h a1 = *(const v16h*)(wb1 + (size_t)q * 512);
        acc[1] = wmma_f16(a1, b, acc[1]);
      }
    }
  }

  const int p = wT + pixSub * 16 + (lane & 15);
  #pragma unroll
  for (int a = 0; a < ACCS; a++) {
    if (tile0 + a >= tiles) continue;
    #pragma unroll
    for (int e = 0; e < 8; e++) {
      int co = (tile0 + a) * 16 + e + ((lane >= 16) ? 8 : 0);
      if (co < Cout && p < W_) {
        float v = acc[a][e] + bias[co];
        if (addt) v += addt[((size_t)n * Cout + co) * HW_ + (size_t)h * W_ + p];
        if (ACT == 1) v = fmaxf(v, 0.f);
        else if (ACT == 2) v = 1.f / (1.f + __expf(-v));
        else if (ACT == 3) v = tanhf(v);
        out[((size_t)n * outC + outCO + co) * HW_ + (size_t)h * W_ + p] = v * scale;
      }
    }
  }
}

// ---------------------------------------------------------------------------
// Correlation: pyr0[n,h,w1,w2] = (1/16) * sum_c f1[n,c,h,w1]*f2[n,c,h,w2]
// Per block: one (n,h), 128x64 output tile; 8 waves x (2x2) WMMA tiles each.
// ---------------------------------------------------------------------------
__global__ __launch_bounds__(256) void corr_gemm_k(const float* __restrict__ f1,
                                                   const float* __restrict__ f2,
                                                   float* __restrict__ pyr0) {
  __shared__ alignas(32) _Float16 sA[128 * 32];
  __shared__ alignas(32) _Float16 sB[64 * 32];
  const int h = blockIdx.y, n = blockIdx.z;
  const int w1T = (blockIdx.x >> 2) * 128;
  const int w2T = (blockIdx.x & 3) * 64;
  const int tid = threadIdx.x, lane = tid & 31, wave = tid >> 5;
  const int w1p = (wave & 3) * 2;   // two w1 subtiles: w1p, w1p+1 (of 8)
  const int w2p = (wave >> 2) * 2;  // two w2 subtiles: w2p, w2p+1 (of 4)

  v8f acc[2][2];
  #pragma unroll
  for (int s = 0; s < 2; s++)
    #pragma unroll
    for (int t = 0; t < 2; t++) acc[s][t] = (v8f){0.f,0.f,0.f,0.f,0.f,0.f,0.f,0.f};

  for (int ck = 0; ck < C_ / 32; ck++) {
    __syncthreads();
    for (int i = tid; i < (128 + 64) * 32 / 4; i += 256) {
      int base4 = i * 4;
      bool isB = base4 >= 128 * 32;
      int b = isB ? base4 - 128 * 32 : base4;
      int x  = b >> 5;
      int c0 = b & 31;
      int gw = (isB ? w2T : w1T) + x;
      union { _Float16 hh[4]; unsigned long long u; } p;
      p.u = 0ull;
      if (gw < W_) {
        const float* src = isB ? f2 : f1;
        size_t gbase = (((size_t)n * C_ + (ck * 32 + c0)) * H_ + h) * W_ + gw;
        #pragma unroll
        for (int j = 0; j < 4; j++) p.hh[j] = (_Float16)src[gbase + (size_t)j * HW_];
      }
      *(unsigned long long*)&(isB ? sB : sA)[x * 32 + c0] = p.u;
    }
    __syncthreads();

    const int cb16 = (lane < 16) ? 0 : 8;
    union { v16h v; v8h h2[2]; } a[2];
    #pragma unroll
    for (int s = 0; s < 2; s++) {
      const _Float16* ap = &sA[((w1p + s) * 16 + (lane & 15)) * 32 + ((lane < 16) ? 0 : 8)];
      a[s].h2[0] = *(const v8h*)ap;
      a[s].h2[1] = *(const v8h*)(ap + 16);
    }
    #pragma unroll
    for (int t = 0; t < 2; t++) {
      union { v16h v; v8h h2[2]; } bu;
      bu.h2[0] = *(const v8h*)&sB[((w2p + t) * 16 + (lane & 15)) * 32 + cb16];
      bu.h2[1] = *(const v8h*)&sB[((w2p + t) * 16 + (lane & 15)) * 32 + cb16 + 16];
      v16h b = bu.v;
      acc[0][t] = wmma_f16(a[0].v, b, acc[0][t]);
      acc[1][t] = wmma_f16(a[1].v, b, acc[1][t]);
    }
  }

  #pragma unroll
  for (int s = 0; s < 2; s++) {
    #pragma unroll
    for (int e = 0; e < 8; e++) {
      int w1 = w1T + (w1p + s) * 16 + e + ((lane >= 16) ? 8 : 0);
      if (w1 >= W_) continue;
      size_t base = (((size_t)n * H_ + h) * W_ + w1) * (size_t)W_;
      #pragma unroll
      for (int t = 0; t < 2; t++) {
        int w2 = w2T + (w2p + t) * 16 + (lane & 15);
        if (w2 < W_) pyr0[base + w2] = acc[s][t][e] * 0.0625f;
      }
    }
  }
}

// Pyramid downsample along last axis (pairwise mean).
__global__ void down_k(const float* __restrict__ in, float* __restrict__ out,
                       int wl_in, int total) {
  int i = blockIdx.x * blockDim.x + threadIdx.x;
  if (i >= total) return;
  int wl_out = wl_in >> 1;
  int w2 = i % wl_out;
  int r  = i / wl_out;
  out[i] = 0.5f * (in[(size_t)r * wl_in + 2 * w2] + in[(size_t)r * wl_in + 2 * w2 + 1]);
}

// Correlation lookup with linear interpolation (zero outside).
__global__ void corr_lookup_k(const float* __restrict__ p0, const float* __restrict__ p1,
                              const float* __restrict__ p2, const float* __restrict__ p3,
                              const float* __restrict__ flow, float* __restrict__ corr) {
  int i = blockIdx.x * blockDim.x + threadIdx.x;
  if (i >= N_ * H_ * W_ * 4) return;
  int l = i & 3;
  int rest = i >> 2;
  int w = rest % W_;
  int rest2 = rest / W_;
  int h = rest2 % H_;
  int n = rest2 / H_;
  const float* pl = (l == 0) ? p0 : (l == 1) ? p1 : (l == 2) ? p2 : p3;
  int wl = W_ >> l;
  float cx = (float)w + flow[((size_t)n * 2) * HW_ + (size_t)h * W_ + w];
  float xs0 = cx / (float)(1 << l);
  size_t base = (((size_t)n * H_ + h) * W_ + w) * (size_t)wl;
  for (int j = 0; j < 9; j++) {
    float xs = xs0 + (float)(j - 4);
    float x0 = floorf(xs);
    float f  = xs - x0;
    int   i0 = (int)x0;
    float v0 = ((unsigned)i0       < (unsigned)wl) ? pl[base + i0]     : 0.f;
    float v1 = ((unsigned)(i0 + 1) < (unsigned)wl) ? pl[base + i0 + 1] : 0.f;
    corr[(((size_t)n * CORRCH_ + l * 9 + j) * H_ + h) * W_ + w] = (1.f - f) * v0 + f * v1;
  }
}

// Elementwise helpers -------------------------------------------------------
__global__ void tanh_k(const float* a, float* o, int n) {
  int i = blockIdx.x * blockDim.x + threadIdx.x;
  if (i < n) o[i] = tanhf(a[i]);
}
__global__ void zero_k(float* o, int n) {
  int i = blockIdx.x * blockDim.x + threadIdx.x;
  if (i < n) o[i] = 0.f;
}
__global__ void mul_inplace_k(float* r, const float* net, int n) {
  int i = blockIdx.x * blockDim.x + threadIdx.x;
  if (i < n) r[i] *= net[i];
}
__global__ void gru_update_k(float* net, const float* z, const float* q, int n) {
  int i = blockIdx.x * blockDim.x + threadIdx.x;
  if (i < n) { float zz = z[i]; net[i] = (1.f - zz) * net[i] + zz * q[i]; }
}
__global__ void copy_flow_k(const float* flow, float* motion) {
  int i = blockIdx.x * blockDim.x + threadIdx.x;
  if (i >= N_ * 2 * (int)HW_) return;
  int p = i % (int)HW_;
  int c = (i / (int)HW_) % 2;
  int n = i / (2 * (int)HW_);
  motion[((size_t)n * HID_ + 126 + c) * HW_ + p] = flow[((size_t)n * 2 + c) * HW_ + p];
}
__global__ void flow_update_k(float* flow, const float* delta) {
  int i = blockIdx.x * blockDim.x + threadIdx.x;
  if (i >= N_ * (int)HW_) return;
  int p = i % (int)HW_;
  int n = i / (int)HW_;
  flow[((size_t)n * 2) * HW_ + p] += delta[((size_t)n * 2) * HW_ + p];  // x only; dy forced 0
}

// Convex-combination 1D upsample (F.unfold [1,9] pattern), writes one iter slice.
__global__ void upsample_k(const float* __restrict__ flow, const float* __restrict__ mask,
                           float* __restrict__ out) {
  int i = blockIdx.x * blockDim.x + threadIdx.x;
  const int OW = 4 * W_, OH = 4 * H_;
  if (i >= N_ * OH * OW) return;
  int ow = i % OW;
  int t  = i / OW;
  int oh = t % OH;
  int n  = t / OH;
  int w = ow >> 2, fx = ow & 3, h = oh >> 2, fy = oh & 3;
  float m[9], mx = -1e30f;
  for (int k = 0; k < 9; k++) {
    m[k] = mask[(((size_t)n * 144 + k * 16 + fy * 4 + fx) * H_ + h) * W_ + w];
    mx = fmaxf(mx, m[k]);
  }
  float s = 0.f;
  for (int k = 0; k < 9; k++) { m[k] = __expf(m[k] - mx); s += m[k]; }
  float acc = 0.f;
  for (int k = 0; k < 9; k++) {
    int wi = w + k - 4;
    float fv = ((unsigned)wi < (unsigned)W_)
                   ? flow[((size_t)n * 2) * HW_ + (size_t)h * W_ + wi] : 0.f;
    acc += m[k] * 4.f * fv;
  }
  out[((size_t)n * OH + oh) * OW + ow] = acc / s;
}

// ---------------------------------------------------------------------------
static inline int cdiv(int a, int b) { return (a + b - 1) / b; }

extern "C" void kernel_launch(void* const* d_in, const int* in_sizes, int n_in,
                              void* d_out, int out_size, void* d_ws, size_t ws_size,
                              hipStream_t stream) {
  const float* fmap1 = (const float*)d_in[0];
  const float* fmap2 = (const float*)d_in[1];
  const float* net0  = (const float*)d_in[2];
  const float* cz    = (const float*)d_in[3];
  const float* cr    = (const float*)d_in[4];
  const float* cq    = (const float*)d_in[5];
  const float* c1_w = (const float*)d_in[6];  const float* c1_b = (const float*)d_in[7];
  const float* c2_w = (const float*)d_in[8];  const float* c2_b = (const float*)d_in[9];
  const float* f1_w = (const float*)d_in[10]; const float* f1_b = (const float*)d_in[11];
  const float* f2_w = (const float*)d_in[12]; const float* f2_b = (const float*)d_in[13];
  const float* cm_w = (const float*)d_in[14]; const float* cm_b = (const float*)d_in[15];
  const float* gz_w = (const float*)d_in[16]; const float* gz_b = (const float*)d_in[17];
  const float* gr_w = (const float*)d_in[18]; const float* gr_b = (const float*)d_in[19];
  const float* gq_w = (const float*)d_in[20]; const float* gq_b = (const float*)d_in[21];
  const float* fh1_w = (const float*)d_in[22]; const float* fh1_b = (const float*)d_in[23];
  const float* fh2_w = (const float*)d_in[24]; const float* fh2_b = (const float*)d_in[25];
  const float* mk1_w = (const float*)d_in[26]; const float* mk1_b = (const float*)d_in[27];
  const float* mk2_w = (const float*)d_in[28]; const float* mk2_b = (const float*)d_in[29];
  float* out = (float*)d_out;

  // ---- workspace carve-out (deterministic) ----
  size_t off = 0;
  auto allocF = [&](size_t nf) {
    float* p = (float*)((char*)d_ws + off);
    off += ((nf * sizeof(float) + 255) & ~(size_t)255);
    return p;
  };
  auto allocH = [&](size_t nh) {
    _Float16* p = (_Float16*)((char*)d_ws + off);
    off += ((nh * sizeof(_Float16) + 255) & ~(size_t)255);
    return p;
  };
  float* pyr0 = allocF((size_t)N_ * H_ * W_ * 240);
  float* pyr1 = allocF((size_t)N_ * H_ * W_ * 120);
  float* pyr2 = allocF((size_t)N_ * H_ * W_ * 60);
  float* pyr3 = allocF((size_t)N_ * H_ * W_ * 30);
  float* net   = allocF((size_t)N_ * HID_ * HW_);
  float* flow  = allocF((size_t)N_ * 2 * HW_);
  float* corr  = allocF((size_t)N_ * CORRCH_ * HW_);
  float* t64   = allocF((size_t)N_ * 64 * HW_);
  float* cat128= allocF((size_t)N_ * 128 * HW_);   // also reused as q
  float* motion= allocF((size_t)N_ * 128 * HW_);
  float* zb    = allocF((size_t)N_ * 128 * HW_);
  float* rb    = allocF((size_t)N_ * 128 * HW_);   // reused as r*net
  float* t256  = allocF((size_t)N_ * 256 * HW_);
  float* delta = allocF((size_t)N_ * 2 * HW_);
  float* maskb = allocF((size_t)N_ * 144 * HW_);

  auto packSize = [](int Cout, int Cin, int khw) {
    return (size_t)cdiv(Cout, 16) * cdiv(Cin, 32) * khw * 512;
  };
  struct PW { const float* w; int Cout, Cin, khw; _Float16* pk; };
  PW pws[12] = {
    {c1_w, 64, 36, 1, nullptr},  {c2_w, 64, 64, 9, nullptr},
    {f1_w, 64, 2, 49, nullptr},  {f2_w, 64, 64, 9, nullptr},
    {cm_w, 126, 128, 9, nullptr},{gz_w, 128, 256, 9, nullptr},
    {gr_w, 128, 256, 9, nullptr},{gq_w, 128, 256, 9, nullptr},
    {fh1_w, 256, 128, 9, nullptr},{fh2_w, 2, 256, 9, nullptr},
    {mk1_w, 256, 128, 9, nullptr},{mk2_w, 144, 256, 1, nullptr},
  };
  for (int i = 0; i < 12; i++) pws[i].pk = allocH(packSize(pws[i].Cout, pws[i].Cin, pws[i].khw));

  // ---- pack weights ----
  for (int i = 0; i < 12; i++) {
    int total = (int)packSize(pws[i].Cout, pws[i].Cin, pws[i].khw);
    pack_w_k<<<cdiv(total, 256), 256, 0, stream>>>(pws[i].w, pws[i].pk,
                                                   pws[i].Cout, pws[i].Cin, pws[i].khw, total);
  }
  _Float16* pk_c1 = pws[0].pk; _Float16* pk_c2 = pws[1].pk;
  _Float16* pk_f1 = pws[2].pk; _Float16* pk_f2 = pws[3].pk;
  _Float16* pk_cm = pws[4].pk; _Float16* pk_gz = pws[5].pk;
  _Float16* pk_gr = pws[6].pk; _Float16* pk_gq = pws[7].pk;
  _Float16* pk_fh1 = pws[8].pk; _Float16* pk_fh2 = pws[9].pk;
  _Float16* pk_mk1 = pws[10].pk; _Float16* pk_mk2 = pws[11].pk;

  // ---- correlation pyramid ----
  corr_gemm_k<<<dim3(8, H_, N_), 256, 0, stream>>>(fmap1, fmap2, pyr0);
  {
    int t1 = N_ * H_ * W_ * 120;
    down_k<<<cdiv(t1, 256), 256, 0, stream>>>(pyr0, pyr1, 240, t1);
    int t2 = N_ * H_ * W_ * 60;
    down_k<<<cdiv(t2, 256), 256, 0, stream>>>(pyr1, pyr2, 120, t2);
    int t3 = N_ * H_ * W_ * 30;
    down_k<<<cdiv(t3, 256), 256, 0, stream>>>(pyr2, pyr3, 60, t3);
  }

  // ---- init ----
  {
    int nh = N_ * HID_ * (int)HW_;
    tanh_k<<<cdiv(nh, 256), 256, 0, stream>>>(net0, net, nh);
    int nf = N_ * 2 * (int)HW_;
    zero_k<<<cdiv(nf, 256), 256, 0, stream>>>(flow, nf);
  }

  const int nHID = N_ * HID_ * (int)HW_;
  auto g128 = [](int Cout) { return dim3(cdiv(W_, 128), H_, N_ * cdiv(Cout, 32)); };
  auto g64  = [](int Cout) { return dim3(cdiv(W_, 64),  H_, N_ * cdiv(Cout, 32)); };

  for (int it = 0; it < ITERS_; it++) {
    // corr lookup
    {
      int t = N_ * H_ * W_ * 4;
      corr_lookup_k<<<cdiv(t, 256), 256, 0, stream>>>(pyr0, pyr1, pyr2, pyr3, flow, corr);
    }
    // motion encoder
    conv_wmma_k<1,1,1,128><<<g128(64), 256, 0, stream>>>(corr, 36, nullptr, 36,
        pk_c1, c1_b, nullptr, t64, 64, 0, 64, 1.f);
    conv_wmma_k<3,3,1,128><<<g128(64), 256, 0, stream>>>(t64, 64, nullptr, 64,
        pk_c2, c2_b, nullptr, cat128, 64, 0, 128, 1.f);
    conv_wmma_k<7,7,1,64><<<g64(64), 256, 0, stream>>>(flow, 2, nullptr, 2,
        pk_f1, f1_b, nullptr, t64, 64, 0, 64, 1.f);
    conv_wmma_k<3,3,1,128><<<g128(64), 256, 0, stream>>>(t64, 64, nullptr, 64,
        pk_f2, f2_b, nullptr, cat128, 64, 64, 128, 1.f);
    conv_wmma_k<3,3,1,128><<<g128(126), 256, 0, stream>>>(cat128, 128, nullptr, 128,
        pk_cm, cm_b, nullptr, motion, 126, 0, 128, 1.f);
    {
      int t = N_ * 2 * (int)HW_;
      copy_flow_k<<<cdiv(t, 256), 256, 0, stream>>>(flow, motion);
    }
    // ConvGRU (hx = net || motion via channel-split inputs)
    conv_wmma_k<3,3,2,128><<<g128(128), 256, 0, stream>>>(net, 128, motion, 256,
        pk_gz, gz_b, cz, zb, 128, 0, 128, 1.f);
    conv_wmma_k<3,3,2,128><<<g128(128), 256, 0, stream>>>(net, 128, motion, 256,
        pk_gr, gr_b, cr, rb, 128, 0, 128, 1.f);
    mul_inplace_k<<<cdiv(nHID, 256), 256, 0, stream>>>(rb, net, nHID);
    conv_wmma_k<3,3,3,128><<<g128(128), 256, 0, stream>>>(rb, 128, motion, 256,
        pk_gq, gq_b, cq, cat128, 128, 0, 128, 1.f);       // q -> cat128 (reuse)
    gru_update_k<<<cdiv(nHID, 256), 256, 0, stream>>>(net, zb, cat128, nHID);
    // flow head
    conv_wmma_k<3,3,1,128><<<g128(256), 256, 0, stream>>>(net, 128, nullptr, 128,
        pk_fh1, fh1_b, nullptr, t256, 256, 0, 256, 1.f);
    conv_wmma_k<3,3,0,128><<<g128(2), 256, 0, stream>>>(t256, 256, nullptr, 256,
        pk_fh2, fh2_b, nullptr, delta, 2, 0, 2, 1.f);
    // mask head
    conv_wmma_k<3,3,1,128><<<g128(256), 256, 0, stream>>>(net, 128, nullptr, 128,
        pk_mk1, mk1_b, nullptr, t256, 256, 0, 256, 1.f);
    conv_wmma_k<1,1,0,128><<<g128(144), 256, 0, stream>>>(t256, 256, nullptr, 256,
        pk_mk2, mk2_b, nullptr, maskb, 144, 0, 144, 0.25f);
    // coords update (dy forced to zero) + upsampled prediction
    {
      int t = N_ * (int)HW_;
      flow_update_k<<<cdiv(t, 256), 256, 0, stream>>>(flow, delta);
      int tu = N_ * 4 * H_ * 4 * W_;
      upsample_k<<<cdiv(tu, 256), 256, 0, stream>>>(
          flow, maskb, out + (size_t)it * N_ * 4 * H_ * 4 * W_);
    }
  }
}
